// SupervisedContrastiveLoss_45603962749034
// MI455X (gfx1250) — compile-verified, matrix operand order checked by reading, not run on hardware
//
#include <hip/hip_runtime.h>
#include <hip/hip_bf16.h>

typedef __bf16 bf16_t;
typedef __attribute__((ext_vector_type(16))) __bf16 v16bf;
typedef __attribute__((ext_vector_type(8)))  __bf16 v8bf;
typedef __attribute__((ext_vector_type(8)))  float   v8f;
typedef __attribute__((ext_vector_type(4)))  float   v4f;

#define SCL_B   8192
#define SCL_D   256
#define SCL_NJ  32                      // column chunks per row strip
#define SCL_JCHUNK (SCL_B / SCL_NJ)     // 256 columns per chunk
#define SCL_INV_T (1.0f / 0.07f)
#define SCL_MFIX  16.0f                 // fixed logsumexp shift; masked <= 14.2857+eps
#define SCL_LOG_EPS (-18.420680743952367f)  // log(1e-8)
#define SCL_RED_BLOCKS 32               // stage-A reduction blocks

// ---------------------------------------------------------------------------
// Kernel 1: L2-normalize each row, split each value into bf16 hi + bf16 lo.
// One wave (32 lanes) per row; each lane owns 8 contiguous floats.
// ---------------------------------------------------------------------------
__global__ __launch_bounds__(256)
void scl_normalize_kernel(const float* __restrict__ hidden,
                          bf16_t* __restrict__ hhi,
                          bf16_t* __restrict__ hlo) {
    const int wave = threadIdx.x >> 5;
    const int lane = threadIdx.x & 31;
    const int row  = blockIdx.x * 8 + wave;
    const float* rp = hidden + (size_t)row * SCL_D;

    v4f a = *(const v4f*)(rp + lane * 8);
    v4f b = *(const v4f*)(rp + lane * 8 + 4);

    float ss = a.x*a.x + a.y*a.y + a.z*a.z + a.w*a.w
             + b.x*b.x + b.y*b.y + b.z*b.z + b.w*b.w;
    #pragma unroll
    for (int m = 16; m >= 1; m >>= 1) ss += __shfl_xor(ss, m, 32);

    float nrm = sqrtf(ss);
    nrm = fmaxf(nrm, 1e-12f);
    const float sc = 1.0f / nrm;

    float v[8] = { a.x*sc, a.y*sc, a.z*sc, a.w*sc,
                   b.x*sc, b.y*sc, b.z*sc, b.w*sc };
    v8bf hi8, lo8;
    #pragma unroll
    for (int e = 0; e < 8; ++e) {
        bf16_t h = (bf16_t)v[e];
        float  r = v[e] - (float)h;
        hi8[e] = h;
        lo8[e] = (bf16_t)r;
    }
    *(v8bf*)(hhi + (size_t)row * SCL_D + lane * 8) = hi8;
    *(v8bf*)(hlo + (size_t)row * SCL_D + lane * 8) = lo8;
}

// ---------------------------------------------------------------------------
// Kernel 2: fused sim-tile GEMM (split-bf16 WMMA) + masked-logsumexp partials.
// grid = (B/64, NJ); block = 4 waves. The 4 waves of a workgroup cover 4
// consecutive 16-row strips with the SAME column chunk, so the B tiles they
// stream are shared through the WGP$ (4x less L2 traffic).
//
// Fragment layouts (ISA 7.12.2, 16-bit 16x32 A / 32x16 B, wave32):
//   lane l: r = l & 15, kb = (l < 16) ? 0 : 8
//   A: row (i0+r), elements K = k0+kb+0..7 then k0+kb+16..23
//   B: col (j0+r) == row (j0+r) of Hn (B = Hn^T), same K packing
// C/D: lane l holds N = l&15; VGPR v holds M = (l<16?0:8)+v.
// ---------------------------------------------------------------------------
__global__ __launch_bounds__(128)
void scl_simloss_kernel(const bf16_t* __restrict__ hhi,
                        const bf16_t* __restrict__ hlo,
                        const int* __restrict__ labels,
                        v4f* __restrict__ partials) {
    const int lane  = threadIdx.x & 31;
    const int wave  = threadIdx.x >> 5;
    const int l15   = lane & 15;
    const int kb    = (lane < 16) ? 0 : 8;
    const int i0    = (blockIdx.x * 4 + wave) * 16;
    const int chunk = blockIdx.y;
    const int j0b   = chunk * SCL_JCHUNK;

    // Preload A fragments (16 rows x 256 K) for hi and lo, 8 K-steps of 32.
    v16bf a_hi[8], a_lo[8];
    {
        const bf16_t* arh = hhi + (size_t)(i0 + l15) * SCL_D;
        const bf16_t* arl = hlo + (size_t)(i0 + l15) * SCL_D;
        #pragma unroll
        for (int ks = 0; ks < 8; ++ks) {
            const int k0 = ks * 32 + kb;
            v8bf h0 = *(const v8bf*)(arh + k0);
            v8bf h1 = *(const v8bf*)(arh + k0 + 16);
            a_hi[ks] = __builtin_shufflevector(h0, h1,
                0,1,2,3,4,5,6,7,8,9,10,11,12,13,14,15);
            v8bf l0 = *(const v8bf*)(arl + k0);
            v8bf l1 = *(const v8bf*)(arl + k0 + 16);
            a_lo[ks] = __builtin_shufflevector(l0, l1,
                0,1,2,3,4,5,6,7,8,9,10,11,12,13,14,15);
        }
    }

    int lab_i[8];
    #pragma unroll
    for (int v = 0; v < 8; ++v) lab_i[v] = labels[i0 + kb + v];

    float s[8], ps[8], cnt[8];
    #pragma unroll
    for (int v = 0; v < 8; ++v) { s[v] = 0.0f; ps[v] = 0.0f; cnt[v] = 0.0f; }

    for (int jt = 0; jt < SCL_JCHUNK / 16; ++jt) {
        const int j0 = j0b + jt * 16;
        const bf16_t* brh = hhi + (size_t)(j0 + l15) * SCL_D;
        const bf16_t* brl = hlo + (size_t)(j0 + l15) * SCL_D;

        v8f c = {};
        #pragma unroll
        for (int ks = 0; ks < 8; ++ks) {
            const int k0 = ks * 32 + kb;
            v8bf b0 = *(const v8bf*)(brh + k0);
            v8bf b1 = *(const v8bf*)(brh + k0 + 16);
            v16bf bh = __builtin_shufflevector(b0, b1,
                0,1,2,3,4,5,6,7,8,9,10,11,12,13,14,15);
            v8bf c0 = *(const v8bf*)(brl + k0);
            v8bf c1 = *(const v8bf*)(brl + k0 + 16);
            v16bf bl = __builtin_shufflevector(c0, c1,
                0,1,2,3,4,5,6,7,8,9,10,11,12,13,14,15);
            // x*y ~= hi*hi + hi*lo + lo*hi  (fp32 accumulate)
            c = __builtin_amdgcn_wmma_f32_16x16x32_bf16(
                    false, a_hi[ks], false, bh, (short)0, c, false, false);
            c = __builtin_amdgcn_wmma_f32_16x16x32_bf16(
                    false, a_hi[ks], false, bl, (short)0, c, false, false);
            c = __builtin_amdgcn_wmma_f32_16x16x32_bf16(
                    false, a_lo[ks], false, bh, (short)0, c, false, false);
        }

        const int jcol  = j0 + l15;
        const int lab_j = labels[jcol];
        #pragma unroll
        for (int v = 0; v < 8; ++v) {
            const int irow = i0 + kb + v;
            const float sim = c[v] * SCL_INV_T;
            // masked = sim + log((1-eye)+1e-8): diag -> log(1e-8), else +~1e-8
            const float masked = sim + ((jcol == irow) ? SCL_LOG_EPS : 1e-8f);
            s[v] += __expf(masked - SCL_MFIX);
            const bool pos = (lab_j == lab_i[v]) && (jcol != irow);
            ps[v]  += pos ? sim : 0.0f;
            cnt[v] += pos ? 1.0f : 0.0f;
        }
    }

    // Reduce across the 16 lanes sharing each output row.
    #pragma unroll
    for (int v = 0; v < 8; ++v) {
        #pragma unroll
        for (int m = 1; m < 16; m <<= 1) {
            s[v]   += __shfl_xor(s[v],   m, 32);
            ps[v]  += __shfl_xor(ps[v],  m, 32);
            cnt[v] += __shfl_xor(cnt[v], m, 32);
        }
    }
    if (l15 == 0) {  // lane 0 -> rows i0..i0+7, lane 16 -> rows i0+8..i0+15
        #pragma unroll
        for (int v = 0; v < 8; ++v) {
            v4f p; p.x = s[v]; p.y = ps[v]; p.z = cnt[v]; p.w = 0.0f;
            partials[(size_t)(i0 + kb + v) * SCL_NJ + chunk] = p;
        }
    }
}

// ---------------------------------------------------------------------------
// Kernel 3a: per-row loss from chunk partials; block-level (loss, valid) sums.
// 32 blocks x 256 threads; one row per thread.
// ---------------------------------------------------------------------------
__global__ __launch_bounds__(256)
void scl_rowloss_kernel(const v4f* __restrict__ partials,
                        float* __restrict__ blockpart) {  // [SCL_RED_BLOCKS][2]
    __shared__ float red_l[8];
    __shared__ float red_v[8];
    const int t    = threadIdx.x;
    const int lane = t & 31;
    const int wave = t >> 5;
    const int row  = blockIdx.x * 256 + t;

    float s = 0.0f, ps = 0.0f, cnt = 0.0f;
    #pragma unroll 4
    for (int c = 0; c < SCL_NJ; ++c) {
        v4f p = partials[(size_t)row * SCL_NJ + c];
        s += p.x; ps += p.y; cnt += p.z;
    }
    const float denom  = SCL_MFIX + __logf(s);
    const float np     = fmaxf(cnt, 1.0f);
    const float valid  = (cnt > 0.0f) ? 1.0f : 0.0f;
    float accL = (-(ps - cnt * denom) / np) * valid;
    float accV = valid;

    #pragma unroll
    for (int m = 16; m >= 1; m >>= 1) {
        accL += __shfl_xor(accL, m, 32);
        accV += __shfl_xor(accV, m, 32);
    }
    if (lane == 0) { red_l[wave] = accL; red_v[wave] = accV; }
    __syncthreads();
    if (t < 8) {
        accL = red_l[t];
        accV = red_v[t];
        #pragma unroll
        for (int m = 4; m >= 1; m >>= 1) {
            accL += __shfl_xor(accL, m, 32);
            accV += __shfl_xor(accV, m, 32);
        }
        if (t == 0) {
            blockpart[blockIdx.x * 2 + 0] = accL;
            blockpart[blockIdx.x * 2 + 1] = accV;
        }
    }
}

// ---------------------------------------------------------------------------
// Kernel 3b: final scalar from the 32 block partials.
// ---------------------------------------------------------------------------
__global__ __launch_bounds__(32)
void scl_final_kernel(const float* __restrict__ blockpart,
                      float* __restrict__ out) {
    const int lane = threadIdx.x & 31;
    float accL = blockpart[lane * 2 + 0];
    float accV = blockpart[lane * 2 + 1];
    #pragma unroll
    for (int m = 16; m >= 1; m >>= 1) {
        accL += __shfl_xor(accL, m, 32);
        accV += __shfl_xor(accV, m, 32);
    }
    if (lane == 0) out[0] = accL / fmaxf(accV, 1.0f);  // WEIGHT = 1.0
}

// ---------------------------------------------------------------------------
extern "C" void kernel_launch(void* const* d_in, const int* in_sizes, int n_in,
                              void* d_out, int out_size, void* d_ws, size_t ws_size,
                              hipStream_t stream) {
    (void)in_sizes; (void)n_in; (void)out_size; (void)ws_size;
    const float* hidden = (const float*)d_in[0];
    const int*   labels = (const int*)d_in[1];
    float* out = (float*)d_out;

    bf16_t* hhi = (bf16_t*)d_ws;                              // 4 MB
    bf16_t* hlo = hhi + (size_t)SCL_B * SCL_D;                // 4 MB
    v4f* partials = (v4f*)(hlo + (size_t)SCL_B * SCL_D);      // 4 MB
    float* blockpart = (float*)(partials + (size_t)SCL_B * SCL_NJ);  // 256 B

    scl_normalize_kernel<<<SCL_B / 8, 256, 0, stream>>>(hidden, hhi, hlo);
    scl_simloss_kernel<<<dim3(SCL_B / 64, SCL_NJ), 128, 0, stream>>>(
        hhi, hlo, labels, partials);
    scl_rowloss_kernel<<<SCL_RED_BLOCKS, 256, 0, stream>>>(partials, blockpart);
    scl_final_kernel<<<1, 32, 0, stream>>>(blockpart, out);
}
